// MultiScaleGraphAttention_23218593202445
// MI455X (gfx1250) — compile-verified, matrix operand order checked by reading, not run on hardware
//
#include <hip/hip_runtime.h>
#include <hip/hip_bf16.h>
#include <cstdint>

// ---------------------------------------------------------------------------
// MultiScaleGraphAttention for MI455X (gfx1250, wave32, WMMA)
//
//  * All big GEMMs run v_wmma_f32_16x16x32_f16: wave computes a 32x64 tile
//    (2 A-frags x 4 B-frags -> 8 WMMA per 12 b128 loads), double-buffered
//    fragment registers so loads for k+32 overlap the WMMA block for k.
//  * e@We ([800k,256] GEMM) folded algebraically into a 64xH matrix.
//  * Edge softmax/scatter in exact f32 (float atomic max / atomic add).
//  * MHA: uniform 250-node graphs -> no pad mask; flash one-pass softmax
//    with K/V staged in LDS per (graph, head).
// ---------------------------------------------------------------------------

#define NN 50000
#define NE 800000
#define NG 200
#define NPGc 250
#define HID 256

typedef __attribute__((ext_vector_type(16))) _Float16 v16h;
typedef __attribute__((ext_vector_type(8)))  float    v8f;
typedef _Float16 half8 __attribute__((ext_vector_type(8)));

#define FLAG_RELU 1
#define FLAG_BIAS 2
#define FLAG_RES  4

__device__ __forceinline__ v16h ld_frag(const _Float16* p) {
  // ISA 16-bit fragment: halves 0..7 = K(kb..kb+7), 8..15 = K(kb+16..kb+23)
  half8 lo = *(const half8*)p;
  half8 hi = *(const half8*)(p + 16);
  v16h f;
#pragma unroll
  for (int j = 0; j < 8; ++j) { f[j] = lo[j]; f[8 + j] = hi[j]; }
  return f;
}

// ---------------------------------------------------------------------------
// WMMA GEMM:  OUT[M,N] = act( A[M,K](f16) @ Bt[N,K]^T(f16) [+bias] [+R] )
// Block: 256 threads = 8 waves -> 256x64 tile. Wave w: rows [w*32, w*32+32).
// ---------------------------------------------------------------------------
__global__ __launch_bounds__(256) void wmma_gemm_f16(
    const _Float16* __restrict__ A, const _Float16* __restrict__ Bt,
    const float* __restrict__ bias, const float* __restrict__ R,
    float* __restrict__ OUT, int M, int N, int K, int flags)
{
  const int lane = threadIdx.x & 31;
  const int wave = threadIdx.x >> 5;
  const int row0 = blockIdx.x * 256 + wave * 32;
  const int col0 = blockIdx.y * 64;
  const int hiHalf = lane >> 4;   // 0: lanes 0-15, 1: lanes 16-31
  const int lr = lane & 15;
  const int kb = hiHalf * 8;

  int arow0 = row0 + lr;       if (arow0 >= M) arow0 = M - 1;  // clamp loads
  int arow1 = row0 + 16 + lr;  if (arow1 >= M) arow1 = M - 1;  // stores guarded

  const _Float16* ap0 = A + (size_t)arow0 * K + kb;
  const _Float16* ap1 = A + (size_t)arow1 * K + kb;
  const _Float16* bp0 = Bt + (size_t)(col0 +  0 + lr) * K + kb;
  const _Float16* bp1 = Bt + (size_t)(col0 + 16 + lr) * K + kb;
  const _Float16* bp2 = Bt + (size_t)(col0 + 32 + lr) * K + kb;
  const _Float16* bp3 = Bt + (size_t)(col0 + 48 + lr) * K + kb;

  v8f acc[2][4] = {};

  // prologue: fragments for k = 0
  v16h a0 = ld_frag(ap0), a1 = ld_frag(ap1);
  v16h b0 = ld_frag(bp0), b1 = ld_frag(bp1), b2 = ld_frag(bp2), b3 = ld_frag(bp3);

  for (int k0 = 0; k0 < K; k0 += 32) {
    // prefetch next k-step (clamped on last iteration: harmless re-load)
    const int kn = (k0 + 32 < K) ? (k0 + 32) : k0;
    v16h na0 = ld_frag(ap0 + kn), na1 = ld_frag(ap1 + kn);
    v16h nb0 = ld_frag(bp0 + kn), nb1 = ld_frag(bp1 + kn);
    v16h nb2 = ld_frag(bp2 + kn), nb3 = ld_frag(bp3 + kn);

    acc[0][0] = __builtin_amdgcn_wmma_f32_16x16x32_f16(false, a0, false, b0, (short)0, acc[0][0], false, false);
    acc[0][1] = __builtin_amdgcn_wmma_f32_16x16x32_f16(false, a0, false, b1, (short)0, acc[0][1], false, false);
    acc[0][2] = __builtin_amdgcn_wmma_f32_16x16x32_f16(false, a0, false, b2, (short)0, acc[0][2], false, false);
    acc[0][3] = __builtin_amdgcn_wmma_f32_16x16x32_f16(false, a0, false, b3, (short)0, acc[0][3], false, false);
    acc[1][0] = __builtin_amdgcn_wmma_f32_16x16x32_f16(false, a1, false, b0, (short)0, acc[1][0], false, false);
    acc[1][1] = __builtin_amdgcn_wmma_f32_16x16x32_f16(false, a1, false, b1, (short)0, acc[1][1], false, false);
    acc[1][2] = __builtin_amdgcn_wmma_f32_16x16x32_f16(false, a1, false, b2, (short)0, acc[1][2], false, false);
    acc[1][3] = __builtin_amdgcn_wmma_f32_16x16x32_f16(false, a1, false, b3, (short)0, acc[1][3], false, false);

    a0 = na0; a1 = na1; b0 = nb0; b1 = nb1; b2 = nb2; b3 = nb3;
  }

#pragma unroll
  for (int r = 0; r < 2; ++r) {
#pragma unroll
    for (int t = 0; t < 4; ++t) {
      const int col = col0 + t * 16 + lr;
#pragma unroll
      for (int v = 0; v < 8; ++v) {
        const int row = row0 + r * 16 + hiHalf * 8 + v;  // D: VGPR v -> M=v (+8)
        if (row < M && col < N) {
          float val = acc[r][t][v];
          if (flags & FLAG_BIAS) val += bias[col];
          if (flags & FLAG_RELU) val = fmaxf(val, 0.f);
          const size_t idx = (size_t)row * N + col;
          if (flags & FLAG_RES) val += R[idx];
          OUT[idx] = val;
        }
      }
    }
  }
}

// --------------------------- small utility kernels -------------------------
__global__ void k_fill(float* p, long long n, float v) {
  long long i = (long long)blockIdx.x * blockDim.x + threadIdx.x;
  if (i < n) p[i] = v;
}

__global__ void k_f32_to_f16(const float* __restrict__ s, _Float16* __restrict__ d, long long n) {
  long long i = (long long)blockIdx.x * blockDim.x + threadIdx.x;
  if (i < n) d[i] = (_Float16)s[i];
}

// W[K,N] row-major -> Bt[N,K] f16 (transposed for contiguous B fragments)
__global__ void k_conv_w_t(const float* __restrict__ W, _Float16* __restrict__ Bt, int K, int N) {
  long long i = (long long)blockIdx.x * blockDim.x + threadIdx.x;
  if (i >= (long long)K * N) return;
  int k = (int)(i / N), n = (int)(i % N);
  Bt[(size_t)n * K + k] = (_Float16)W[i];
}

// generic small dense: OUT[M,N] = act(A@W + bias)
__global__ void k_small_gemm(const float* __restrict__ A, const float* __restrict__ W,
                             const float* __restrict__ bias, float* __restrict__ OUT,
                             int M, int K, int N, int relu) {
  long long i = (long long)blockIdx.x * blockDim.x + threadIdx.x;
  if (i >= (long long)M * N) return;
  int r = (int)(i / N), c = (int)(i % N);
  float s = bias ? bias[c] : 0.f;
  const float* a = A + (size_t)r * K;
  for (int k = 0; k < K; ++k) s += a[k] * W[(size_t)k * N + c];
  if (relu) s = fmaxf(s, 0.f);
  OUT[i] = s;
}

// ------------------------------- GAT pieces --------------------------------
__device__ inline void atomicMaxF(float* a, float v) {
  if (v >= 0.f) atomicMax((int*)a, __float_as_int(v));
  else          atomicMin((unsigned int*)a, __float_as_uint(v));
}

// s_src[n,h], s_dst[n,h] = sum_c hW[n,h*C+c] * att_{src,dst}[h,c]
__global__ void k_att_node(const float* __restrict__ hW, const float* __restrict__ as,
                           const float* __restrict__ ad, float* __restrict__ ssrc,
                           float* __restrict__ sdst, int M, int H, int C) {
  long long i = (long long)blockIdx.x * blockDim.x + threadIdx.x;
  if (i >= (long long)M * H) return;
  int n = (int)(i / H), h = (int)(i % H);
  const float* row = hW + (size_t)n * HID + h * C;
  float a = 0.f, b = 0.f;
  for (int c = 0; c < C; ++c) { a += row[c] * as[h * C + c]; b += row[c] * ad[h * C + c]; }
  ssrc[i] = a; sdst[i] = b;
}

// M[k,h] = sum_c We[k, h*C+c] * att_edge[h,c]   (folds the E x 256 GEMM away)
__global__ void k_make_me(const float* __restrict__ We, const float* __restrict__ ae,
                          float* __restrict__ Me, int H, int C) {
  int i = blockIdx.x * blockDim.x + threadIdx.x;
  if (i >= 64 * H) return;
  int k = i / H, h = i % H;
  float s = 0.f;
  for (int c = 0; c < C; ++c) s += We[(size_t)k * HID + h * C + c] * ae[h * C + c];
  Me[i] = s;
}

__global__ void k_edge_logits(const float* __restrict__ EF, const float* __restrict__ Me,
                              const float* __restrict__ ssrc, const float* __restrict__ sdst,
                              const int* __restrict__ src, const int* __restrict__ dst,
                              float* __restrict__ AE, float* __restrict__ amax, int H) {
  long long i = (long long)blockIdx.x * blockDim.x + threadIdx.x;
  if (i >= (long long)NE * H) return;
  int e = (int)(i / H), h = (int)(i % H);
  const float* er = EF + (size_t)e * 64;
  float s = 0.f;
  for (int k = 0; k < 64; ++k) s += er[k] * Me[k * H + h];
  s += ssrc[(size_t)src[e] * H + h] + sdst[(size_t)dst[e] * H + h];
  s = (s > 0.f) ? s : 0.2f * s;                 // leaky_relu(0.2)
  AE[i] = s;
  atomicMaxF(&amax[(size_t)dst[e] * H + h], s);
}

__global__ void k_edge_exp(float* __restrict__ AE, const float* __restrict__ amax,
                           float* __restrict__ denom, const int* __restrict__ dst, int H) {
  long long i = (long long)blockIdx.x * blockDim.x + threadIdx.x;
  if (i >= (long long)NE * H) return;
  int e = (int)(i / H), h = (int)(i % H);
  float v = __expf(AE[i] - amax[(size_t)dst[e] * H + h]);
  AE[i] = v;
  atomicAdd(&denom[(size_t)dst[e] * H + h], v);
}

// scatter: OUT[dst] += hW[src] * alpha ; 64 threads/edge, 4 channels each
__global__ void k_edge_msg(const float* __restrict__ hW, const float* __restrict__ AE,
                           const float* __restrict__ denom, const int* __restrict__ src,
                           const int* __restrict__ dst, float* __restrict__ OUT, int H, int C) {
  long long i = (long long)blockIdx.x * blockDim.x + threadIdx.x;
  if (i >= (long long)NE * 64) return;
  int e = (int)(i >> 6), t = (int)(i & 63);
  int c0 = t * 4, h = c0 / C;
  int d = dst[e], s = src[e];
  float alpha = AE[(size_t)e * H + h] / (denom[(size_t)d * H + h] + 1e-16f);
  const float* hr = hW + (size_t)s * HID + c0;
  float* orow = OUT + (size_t)d * HID + c0;
#pragma unroll
  for (int j = 0; j < 4; ++j) atomicAdd(&orow[j], hr[j] * alpha);
}

__global__ void k_add_msg_bias(float* __restrict__ X, const float* __restrict__ MSG,
                               const float* __restrict__ bias, long long n) {
  long long i = (long long)blockIdx.x * blockDim.x + threadIdx.x;
  if (i < n) X[i] += MSG[i] + bias[i & (HID - 1)];
}

// ------------------------------- batch norm --------------------------------
__global__ void k_bn_stats(const float* __restrict__ X, int M, int N,
                           float* __restrict__ sums, float* __restrict__ sqs) {
  int c = threadIdx.x;
  if (c >= N) return;
  float s = 0.f, q = 0.f;
  for (int r = blockIdx.x; r < M; r += gridDim.x) {
    float v = X[(size_t)r * N + c]; s += v; q += v * v;
  }
  atomicAdd(&sums[c], s); atomicAdd(&sqs[c], q);
}

__global__ void k_bn_apply(float* __restrict__ X, const float* __restrict__ sums,
                           const float* __restrict__ sqs, const float* __restrict__ gamma,
                           const float* __restrict__ beta, int M, int N, int relu) {
  long long i = (long long)blockIdx.x * blockDim.x + threadIdx.x;
  if (i >= (long long)M * N) return;
  int c = (int)(i % N);
  float mu = sums[c] / (float)M;
  float var = sqs[c] / (float)M - mu * mu;
  float v = gamma[c] * (X[i] - mu) * rsqrtf(var + 1e-5f) + beta[c];
  if (relu) v = fmaxf(v, 0.f);
  X[i] = v;
}

// ------------------------------- MHA ---------------------------------------
// One block per (graph, head). K/V staged in LDS, flash one-pass softmax.
__global__ __launch_bounds__(256) void k_mha_attn(const float* __restrict__ Q,
                                                  const float* __restrict__ Km,
                                                  const float* __restrict__ V,
                                                  float* __restrict__ O) {
  __shared__ float sK[NPGc * 32];
  __shared__ float sV[NPGc * 32];
  const int b = blockIdx.x, h = blockIdx.y;
  const int base = b * NPGc;
  for (int i = threadIdx.x; i < NPGc * 32; i += 256) {
    int r = i >> 5, c = i & 31;
    sK[i] = Km[(size_t)(base + r) * HID + h * 32 + c];
    sV[i] = V[(size_t)(base + r) * HID + h * 32 + c];
  }
  __syncthreads();
  const int qr = threadIdx.x;
  if (qr >= NPGc) return;
  const float scale = 0.17677669529663687f;     // 1/sqrt(32)
  float q[32], acc[32];
#pragma unroll
  for (int c = 0; c < 32; ++c) { q[c] = Q[(size_t)(base + qr) * HID + h * 32 + c] * scale; acc[c] = 0.f; }
  float m = -3.0e38f, l = 0.f;
  for (int j = 0; j < NPGc; ++j) {
    float s = 0.f;
#pragma unroll
    for (int c = 0; c < 32; ++c) s += q[c] * sK[j * 32 + c];
    float mn = fmaxf(m, s);
    float corr = __expf(m - mn);
    float p = __expf(s - mn);
    l = l * corr + p;
#pragma unroll
    for (int c = 0; c < 32; ++c) acc[c] = acc[c] * corr + p * sV[j * 32 + c];
    m = mn;
  }
  float inv = 1.f / l;
#pragma unroll
  for (int c = 0; c < 32; ++c) O[(size_t)(base + qr) * HID + h * 32 + c] = acc[c] * inv;
}

// ---------------------------- pooling / head -------------------------------
__global__ void k_graph_mean(const float* __restrict__ X, float* __restrict__ OUT,
                             int B, int S, int Nc) {
  long long i = (long long)blockIdx.x * blockDim.x + threadIdx.x;
  if (i >= (long long)B * Nc) return;
  int b = (int)(i / Nc), c = (int)(i % Nc);
  const float* base = X + (size_t)b * S * Nc + c;
  float s = 0.f;
  for (int r = 0; r < S; ++r) s += base[(size_t)r * Nc];
  OUT[i] = s / (float)S;
}

__global__ void k_pool(const float* __restrict__ X, const float* __restrict__ w,
                       float* __restrict__ PL) {
  long long i = (long long)blockIdx.x * blockDim.x + threadIdx.x;
  if (i >= (long long)NG * HID) return;
  int b = (int)(i / HID), c = (int)(i % HID);
  const float* base = X + (size_t)b * NPGc * HID + c;
  float s = 0.f, mx = -3.0e38f;
  for (int r = 0; r < NPGc; ++r) { float v = base[(size_t)r * HID]; s += v; mx = fmaxf(mx, v); }
  float mean = s / (float)NPGc;
  PL[i] = (w[0] * mean + w[1] * mx + w[2] * s) * (1.f / 3.f);
}

__global__ void k_concat_final(const float* __restrict__ PL, const float* __restrict__ LM,
                               const float* __restrict__ GF, float* __restrict__ FIN) {
  int i = blockIdx.x * blockDim.x + threadIdx.x;
  if (i >= NG * 384) return;
  int b = i / 384, c = i % 384;
  float v = (c < 256) ? PL[b * 256 + c]
          : (c < 320) ? LM[b * 64 + (c - 256)]
                      : GF[b * 64 + (c - 320)];
  FIN[i] = v;
}

// ---------------------------------------------------------------------------
static inline size_t al256(size_t x) { return (x + 255) & ~(size_t)255; }

extern "C" void kernel_launch(void* const* d_in, const int* in_sizes, int n_in,
                              void* d_out, int out_size, void* d_ws, size_t ws_size,
                              hipStream_t stream) {
  (void)in_sizes; (void)n_in; (void)out_size; (void)ws_size;

  // ---- inputs in setup_inputs() insertion order, params pytree in insertion order
  const float* Xin  = (const float*)d_in[0];
  const int*   EIDX = (const int*)d_in[1];
  const float* EATT = (const float*)d_in[2];
  // d_in[3] = batch (uniform 250 nodes/graph -> layout is implicit)
  const int* srcI = EIDX;
  const int* dstI = EIDX + NE;

  const float* npW = (const float*)d_in[4]; const float* npB = (const float*)d_in[5];
  const float* epW = (const float*)d_in[6]; const float* epB = (const float*)d_in[7];
  const float *lW[4], *lWe[4], *lAs[4], *lAd[4], *lAe[4], *lB[4], *lG[4], *lBe[4],
              *lF1W[4], *lF1B[4], *lF2W[4], *lF2B[4];
  for (int l = 0; l < 4; ++l) {
    int b = 8 + 12 * l;
    lW[l]  = (const float*)d_in[b + 0];  lWe[l] = (const float*)d_in[b + 1];
    lAs[l] = (const float*)d_in[b + 2];  lAd[l] = (const float*)d_in[b + 3];
    lAe[l] = (const float*)d_in[b + 4];  lB[l]  = (const float*)d_in[b + 5];
    lG[l]  = (const float*)d_in[b + 6];  lBe[l] = (const float*)d_in[b + 7];
    lF1W[l] = (const float*)d_in[b + 8]; lF1B[l] = (const float*)d_in[b + 9];
    lF2W[l] = (const float*)d_in[b + 10]; lF2B[l] = (const float*)d_in[b + 11];
  }
  const float* WqW = (const float*)d_in[56]; const float* WqB = (const float*)d_in[57];
  const float* WkW = (const float*)d_in[58]; const float* WkB = (const float*)d_in[59];
  const float* WvW = (const float*)d_in[60]; const float* WvB = (const float*)d_in[61];
  const float* WoW = (const float*)d_in[62]; const float* WoB = (const float*)d_in[63];
  const float* L1W = (const float*)d_in[64]; const float* L1B = (const float*)d_in[65];
  const float* L2W = (const float*)d_in[66]; const float* L2B = (const float*)d_in[67];
  const float* G1W = (const float*)d_in[68]; const float* G1B = (const float*)d_in[69];
  const float* G2W = (const float*)d_in[70]; const float* G2B = (const float*)d_in[71];
  const float* SCW = (const float*)d_in[72];
  const float* O1W = (const float*)d_in[73]; const float* O1B = (const float*)d_in[74];
  const float* OBG = (const float*)d_in[75]; const float* OBB = (const float*)d_in[76];
  const float* O2W = (const float*)d_in[77]; const float* O2B = (const float*)d_in[78];

  // ---- workspace bump allocator (total ~560 MB)
  char* ws = (char*)d_ws; size_t off = 0;
  auto alloc = [&](size_t bytes) -> void* { void* p = ws + off; off = al256(off + bytes); return p; };

  _Float16* npW16 = (_Float16*)alloc((size_t)256 * 128 * 2);
  _Float16 *W16[4], *F116[4], *F216[4];
  for (int l = 0; l < 4; ++l) {
    W16[l]  = (_Float16*)alloc((size_t)256 * 256 * 2);
    F116[l] = (_Float16*)alloc((size_t)512 * 256 * 2);
    F216[l] = (_Float16*)alloc((size_t)256 * 512 * 2);
  }
  _Float16* Wq16 = (_Float16*)alloc((size_t)256 * 256 * 2);
  _Float16* Wk16 = (_Float16*)alloc((size_t)256 * 256 * 2);
  _Float16* Wv16 = (_Float16*)alloc((size_t)256 * 256 * 2);
  _Float16* Wo16 = (_Float16*)alloc((size_t)256 * 256 * 2);
  _Float16* L116 = (_Float16*)alloc((size_t)128 * 256 * 2);
  _Float16* L216 = (_Float16*)alloc((size_t)64 * 128 * 2);

  _Float16* A16 = (_Float16*)alloc((size_t)NN * 512 * 2);           // act staging
  float* X   = (float*)alloc((size_t)NN * HID * 4);                 // node state
  float* TA  = (float*)alloc((size_t)NN * 512 * 4);                 // ff hidden / Q|K
  float* TB  = (float*)alloc((size_t)NN * HID * 4);                 // hW / V / xattn
  float* TC  = (float*)alloc((size_t)NN * HID * 4);                 // msg out / O / local_h
  float* EF  = (float*)alloc((size_t)NE * 64 * 4);                  // edge features
  float* AE  = (float*)alloc((size_t)NE * 8 * 4);                   // logits -> exp
  float* AMX = (float*)alloc((size_t)NN * 8 * 4);
  float* DEN = (float*)alloc((size_t)NN * 8 * 4);
  float* SSR = (float*)alloc((size_t)NN * 8 * 4);
  float* SDS = (float*)alloc((size_t)NN * 8 * 4);
  float* ME  = (float*)alloc((size_t)64 * 8 * 4);
  float* LOC = (float*)alloc((size_t)NN * 64 * 4);
  float* GM  = (float*)alloc((size_t)NG * HID * 4);
  float* G1  = (float*)alloc((size_t)NG * 128 * 4);
  float* GF  = (float*)alloc((size_t)NG * 64 * 4);
  float* PL  = (float*)alloc((size_t)NG * HID * 4);
  float* LM  = (float*)alloc((size_t)NG * 64 * 4);
  float* FIN = (float*)alloc((size_t)NG * 384 * 4);
  float* H1  = (float*)alloc((size_t)NG * HID * 4);
  float* CSUM = (float*)alloc(512 * 4);
  float* CSQ  = (float*)alloc(512 * 4);

  auto blks = [](long long n) { return (unsigned)((n + 255) / 256); };
  auto fill = [&](float* p, long long n, float v) {
    k_fill<<<blks(n), 256, 0, stream>>>(p, n, v);
  };
  auto toh = [&](const float* s, _Float16* d, long long n) {
    k_f32_to_f16<<<blks(n), 256, 0, stream>>>(s, d, n);
  };
  auto convW = [&](const float* W, _Float16* Bt, int K, int N) {
    k_conv_w_t<<<blks((long long)K * N), 256, 0, stream>>>(W, Bt, K, N);
  };
  auto gemm = [&](const _Float16* A, const _Float16* Bt, const float* bias, const float* R,
                  float* OUT, int M, int N, int K, int flags) {
    dim3 g((M + 255) / 256, N / 64);
    wmma_gemm_f16<<<g, 256, 0, stream>>>(A, Bt, bias, R, OUT, M, N, K, flags);
  };

  // ---- stage 0: weight conversion (transposed f16)
  convW(npW, npW16, 128, 256);
  for (int l = 0; l < 4; ++l) {
    convW(lW[l], W16[l], 256, 256);
    convW(lF1W[l], F116[l], 256, 512);
    convW(lF2W[l], F216[l], 512, 256);
  }
  convW(WqW, Wq16, 256, 256); convW(WkW, Wk16, 256, 256);
  convW(WvW, Wv16, 256, 256); convW(WoW, Wo16, 256, 256);
  convW(L1W, L116, 256, 128); convW(L2W, L216, 128, 64);

  // ---- projections
  toh(Xin, A16, (long long)NN * 128);
  gemm(A16, npW16, npB, nullptr, X, NN, 256, 128, FLAG_BIAS);
  k_small_gemm<<<blks((long long)NE * 64), 256, 0, stream>>>(EATT, epW, epB, EF, NE, 16, 64, 0);

  // ---- GAT layers
  const int Hs[4] = {8, 4, 2, 1};
  for (int l = 0; l < 4; ++l) {
    const int H = Hs[l], C = HID / H;
    toh(X, A16, (long long)NN * HID);
    gemm(A16, W16[l], nullptr, nullptr, TB, NN, 256, 256, 0);           // hW

    k_att_node<<<blks((long long)NN * H), 256, 0, stream>>>(TB, lAs[l], lAd[l], SSR, SDS, NN, H, C);
    k_make_me<<<blks(64 * H), 256, 0, stream>>>(lWe[l], lAe[l], ME, H, C);

    fill(AMX, (long long)NN * H, -3.0e38f);
    fill(DEN, (long long)NN * H, 0.f);
    fill(TC, (long long)NN * HID, 0.f);

    k_edge_logits<<<blks((long long)NE * H), 256, 0, stream>>>(EF, ME, SSR, SDS, srcI, dstI, AE, AMX, H);
    k_edge_exp<<<blks((long long)NE * H), 256, 0, stream>>>(AE, AMX, DEN, dstI, H);
    k_edge_msg<<<blks((long long)NE * 64), 256, 0, stream>>>(TB, AE, DEN, srcI, dstI, TC, H, C);

    k_add_msg_bias<<<blks((long long)NN * HID), 256, 0, stream>>>(X, TC, lB[l], (long long)NN * HID);

    fill(CSUM, 256, 0.f); fill(CSQ, 256, 0.f);
    k_bn_stats<<<128, 256, 0, stream>>>(X, NN, 256, CSUM, CSQ);
    k_bn_apply<<<blks((long long)NN * HID), 256, 0, stream>>>(X, CSUM, CSQ, lG[l], lBe[l], NN, 256, 0);

    toh(X, A16, (long long)NN * HID);
    gemm(A16, F116[l], lF1B[l], nullptr, TA, NN, 512, 256, FLAG_BIAS | FLAG_RELU);
    toh(TA, A16, (long long)NN * 512);
    gemm(A16, F216[l], lF2B[l], X, X, NN, 256, 512, FLAG_BIAS | FLAG_RES);  // x += ff
  }

  // ---- MHA (all graphs exactly 250 nodes -> no padding)
  float* Qb = TA;
  float* Kb = TA + (size_t)NN * HID;
  toh(X, A16, (long long)NN * HID);
  gemm(A16, Wq16, WqB, nullptr, Qb, NN, 256, 256, FLAG_BIAS);
  gemm(A16, Wk16, WkB, nullptr, Kb, NN, 256, 256, FLAG_BIAS);
  gemm(A16, Wv16, WvB, nullptr, TB, NN, 256, 256, FLAG_BIAS);
  {
    dim3 g(NG, 8);
    k_mha_attn<<<g, 256, 0, stream>>>(Qb, Kb, TB, TC);
  }
  toh(TC, A16, (long long)NN * HID);
  gemm(A16, Wo16, WoB, nullptr, TB, NN, 256, 256, FLAG_BIAS);             // xattn
  k_graph_mean<<<blks((long long)NG * HID), 256, 0, stream>>>(TB, GM, NG, NPGc, HID);
  k_small_gemm<<<blks((long long)NG * 128), 256, 0, stream>>>(GM, G1W, G1B, G1, NG, 256, 128, 1);
  k_small_gemm<<<blks((long long)NG * 64), 256, 0, stream>>>(G1, G2W, G2B, GF, NG, 128, 64, 0);

  // ---- local MLP + per-graph pooling
  toh(X, A16, (long long)NN * HID);
  gemm(A16, L116, L1B, nullptr, TC, NN, 128, 256, FLAG_BIAS | FLAG_RELU);
  toh(TC, A16, (long long)NN * 128);
  gemm(A16, L216, L2B, nullptr, LOC, NN, 64, 128, FLAG_BIAS);
  k_graph_mean<<<blks((long long)NG * 64), 256, 0, stream>>>(LOC, LM, NG, NPGc, 64);
  k_pool<<<blks((long long)NG * HID), 256, 0, stream>>>(X, SCW, PL);

  // ---- output head
  k_concat_final<<<blks(NG * 384), 256, 0, stream>>>(PL, LM, GF, FIN);
  k_small_gemm<<<blks((long long)NG * HID), 256, 0, stream>>>(FIN, O1W, O1B, H1, NG, 384, 256, 0);
  fill(CSUM, 256, 0.f); fill(CSQ, 256, 0.f);
  k_bn_stats<<<64, 256, 0, stream>>>(H1, NG, 256, CSUM, CSQ);
  k_bn_apply<<<blks((long long)NG * HID), 256, 0, stream>>>(H1, CSUM, CSQ, OBG, OBB, NG, 256, 1);
  k_small_gemm<<<blks((long long)NG * HID), 256, 0, stream>>>(H1, O2W, O2B, (float*)d_out, NG, 256, 256, 0);
}
